// ExactSlidingWindowMHA_23089744183533
// MI455X (gfx1250) — matrix operand order repaired and checked by reading, not compiled, not run on hardware
//
#include <hip/hip_runtime.h>
#include <hip/hip_bf16.h>

// ---------------------------------------------------------------------------
// Sliding-window causal MHA for MI455X (gfx1250), bf16 WMMA pipeline with
// async-to-LDS double-buffered weight staging.
// B=2, S=2048, DIM=1024, H=16, DH=64, WIN=512
// ---------------------------------------------------------------------------

#define B_   2
#define S_   2048
#define DIM_ 1024
#define H_   16
#define DH_  64
#define WIN_ 512
#define M_   (B_ * S_)         // 4096
#define N3_  (3 * H_ * DH_)    // 3072

typedef __attribute__((ext_vector_type(16))) __bf16 v16bf;
typedef __attribute__((ext_vector_type(8)))  float  v8f;
typedef unsigned short ushort_t;
typedef int vsi4 __attribute__((vector_size(16)));   // matches builtin's V4i

#define AS1 __attribute__((address_space(1)))
#define AS3 __attribute__((address_space(3)))

#if defined(__has_builtin)
#  if __has_builtin(__builtin_amdgcn_global_load_async_to_lds_b128)
#    define USE_ASYNC_LDS 1
#  endif
#endif

union Frag {
    uint4    u[2];
    ushort_t s[16];
    v16bf    v;
};

static __device__ __forceinline__ v8f wmma_bf16(v16bf a, v16bf b, v8f c) {
    // D = A(16x32 bf16) * B(32x16 bf16) + C(16x16 f32)
    return __builtin_amdgcn_wmma_f32_16x16x32_bf16(
        /*neg_a=*/false, a, /*neg_b=*/false, b,
        /*c_mod=*/(short)0, c, /*reuse_a=*/false, /*reuse_b=*/false);
}

static __device__ __forceinline__ ushort_t f2bf(float f) {
    unsigned int u = __float_as_uint(f);
    u += 0x7FFFu + ((u >> 16) & 1u);      // round-to-nearest-even
    return (ushort_t)(u >> 16);
}

// 16-byte global -> LDS copy (async on CDNA5 when available)
static __device__ __forceinline__ void stage16(const ushort_t* g, ushort_t* l) {
#if defined(USE_ASYNC_LDS)
    __builtin_amdgcn_global_load_async_to_lds_b128(
        (AS1 vsi4*)(unsigned long long)g,
        (AS3 vsi4*)(unsigned int)(unsigned long long)l,
        /*offset=*/0, /*cpol=*/0);
#else
    *(uint4*)l = *(const uint4*)g;
#endif
}

static __device__ __forceinline__ void wait_stage() {
#if defined(USE_ASYNC_LDS)
#  if __has_builtin(__builtin_amdgcn_s_wait_asynccnt)
    __builtin_amdgcn_s_wait_asynccnt(0);
#  else
    asm volatile("s_wait_asynccnt 0x0" ::: "memory");
#  endif
#endif
    __syncthreads();
}

// ---------------------------------------------------------------------------
// Shared GEMM core: block = 256 threads = 8 waves computing a 128x64 tile of
// C = A(M,1024) @ BT(N,1024)^T. Weight chunk (64 cols x 32 K, bf16) staged in
// LDS, double-buffered, shared by all waves. K fixed at 1024.
// LDS row stride padded to 40 halves (80B) -> conflict-free b128 reads.
// ---------------------------------------------------------------------------

#define BSTR_ 40                       // padded halves per n-row
#define BBUF_ (64 * BSTR_)             // halves per buffer

static __device__ __forceinline__ void gemm_core_128x64(
    const ushort_t* __restrict__ A,    // (M,1024) bf16
    const ushort_t* __restrict__ BT,   // (N,1024) bf16
    int mrow,                          // this wave's 16-row base
    int n0,                            // block's 64-col base
    ushort_t* sb,                      // shared, 2*BBUF_ halves
    v8f acc[4]) {
    const int tid  = threadIdx.x;      // 0..255
    const int lane = tid & 31;
    const int lm   = lane & 15;
    const int kg   = lane >> 4;

    // staging assignment: thread -> (n-row, 8-half chunk)
    const int sn = tid >> 2;                       // 0..63
    const int sk = (tid & 3) * 8;                  // 0,8,16,24
    const ushort_t* gsrc = BT + (size_t)(n0 + sn) * DIM_ + sk;
    ushort_t* l0 = sb + sn * BSTR_ + sk;
    ushort_t* l1 = sb + BBUF_ + sn * BSTR_ + sk;

    const ushort_t* arow = A + (size_t)(mrow + lm) * DIM_;

    stage16(gsrc, l0);                             // preload chunk 0
    wait_stage();

    for (int it = 0; it < DIM_ / 32; ++it) {
        const int kb = it * 32;
        ushort_t* cur = (it & 1) ? (sb + BBUF_) : sb;
        if (it + 1 < DIM_ / 32)
            stage16(gsrc + kb + 32, (it & 1) ? l0 : l1);   // prefetch next

        Frag fa;
        fa.u[0] = *(const uint4*)(arow + kb + kg * 8);
        fa.u[1] = *(const uint4*)(arow + kb + 16 + kg * 8);
        #pragma unroll
        for (int ns = 0; ns < 4; ++ns) {
            const ushort_t* bl = cur + (ns * 16 + lm) * BSTR_ + kg * 8;
            Frag fb;
            fb.u[0] = *(const uint4*)(bl);
            fb.u[1] = *(const uint4*)(bl + 16);
            acc[ns] = wmma_bf16(fa.v, fb.v, acc[ns]);
        }
        if (it + 1 < DIM_ / 32) wait_stage();
    }
}

// ------------------------------ conversions --------------------------------

__global__ void cvt_bf16_kernel(const float* __restrict__ in,
                                ushort_t* __restrict__ out, int n) {
    for (int i = blockIdx.x * blockDim.x + threadIdx.x; i < n;
         i += gridDim.x * blockDim.x)
        out[i] = f2bf(in[i]);
}

// in: (K,N) row-major fp32   ->   out: (N,K) row-major bf16
__global__ void cvt_bf16_t_kernel(const float* __restrict__ in,
                                  ushort_t* __restrict__ out, int K, int N) {
    int total = K * N;
    for (int i = blockIdx.x * blockDim.x + threadIdx.x; i < total;
         i += gridDim.x * blockDim.x) {
        int n = i / K;
        int k = i - n * K;
        out[i] = f2bf(in[(size_t)k * N + n]);
    }
}

// ------------------------------ QKV GEMM -----------------------------------
// Epilogue scatters q (scaled), k to (B,H,S,DH) and v to (B,H,DH,S).

__global__ void __launch_bounds__(256)
gemm_qkv_kernel(const ushort_t* __restrict__ A,
                const ushort_t* __restrict__ BT,
                ushort_t* __restrict__ qb,
                ushort_t* __restrict__ kbuf,
                ushort_t* __restrict__ vb) {
    __shared__ __align__(16) ushort_t sb[2 * BBUF_];
    const int bm   = blockIdx.x & 31;          // M/128 = 32
    const int nt4  = blockIdx.x >> 5;          // N3/64 = 48
    const int lane = threadIdx.x & 31;
    const int lm   = lane & 15;
    const int kg   = lane >> 4;
    const int mrow = bm * 128 + (threadIdx.x >> 5) * 16;
    const int n0   = nt4 * 64;

    v8f acc[4] = {};
    gemm_core_128x64(A, BT, mrow, n0, sb, acc);

    const float qscale = 0.125f;               // DH^-0.5
    #pragma unroll
    for (int ns = 0; ns < 4; ++ns) {
        const int n = n0 + ns * 16 + lm;
        #pragma unroll
        for (int r = 0; r < 8; ++r) {
            const int gm = mrow + r + kg * 8;
            const int b  = gm >> 11;           // / S_
            const int s  = gm & (S_ - 1);
            const float val = acc[ns][r];
            if (n < 1024) {
                const int h = n >> 6, dh = n & 63;
                qb[(((size_t)(b * H_ + h) * S_) + s) * DH_ + dh] =
                    f2bf(val * qscale);
            } else if (n < 2048) {
                const int nn = n - 1024;
                const int h = nn >> 6, dh = nn & 63;
                kbuf[(((size_t)(b * H_ + h) * S_) + s) * DH_ + dh] = f2bf(val);
            } else {
                const int nn = n - 2048;
                const int h = nn >> 6, dh = nn & 63;
                vb[(((size_t)(b * H_ + h) * DH_) + dh) * S_ + s] = f2bf(val);
            }
        }
    }
}

// --------------------------- attention kernel ------------------------------
// One wave (32 threads) per (b,h, 16-query tile). Window span = 33 key tiles.
// LDS scoreboard: 16 rows x 544 cols fp32 (cols 528..543 are zero padding).

#define NTILE_  33
#define SCOLS_  544
#define SVALID_ 528

__global__ void __launch_bounds__(32)
attn_kernel(const ushort_t* __restrict__ qb,
            const ushort_t* __restrict__ kb,
            const ushort_t* __restrict__ vb,
            ushort_t* __restrict__ ab) {
    __shared__ __align__(16) float sc[16 * SCOLS_];
    __shared__ float rowsum[16];

    const int lane = threadIdx.x;
    const int bh = blockIdx.x / (S_ / 16);     // b*H + h
    const int qt = blockIdx.x % (S_ / 16);
    const int b  = bh >> 4;
    const int h  = bh & 15;
    const int i0 = qt * 16;
    const int jbase = i0 - WIN_;               // multiple of 16 (can be <0)

    const int lm = lane & 15;
    const int kg = lane >> 4;

    // init scoreboard: -FLT_MAX in valid zone, 0 in padding zone
    for (int t = lane; t < 16 * SCOLS_; t += 32) {
        const int c = t % SCOLS_;
        sc[t] = (c < SVALID_) ? -3.402823466e38f : 0.0f;
    }
    __syncthreads();

    // q fragments for this query tile (K = DH = 64 -> two 16x32 A-frags)
    const ushort_t* qrow = qb + ((size_t)bh * S_ + i0 + lm) * DH_;
    Frag a0, a1;
    a0.u[0] = *(const uint4*)(qrow + kg * 16);
    a0.u[1] = *(const uint4*)(qrow + kg * 16 + 8);
    a1.u[0] = *(const uint4*)(qrow + 32 + kg * 16);
    a1.u[1] = *(const uint4*)(qrow + 32 + kg * 16 + 8);

    auto load_ktile = [&](int j0, Frag& b0, Frag& b1) {
        const ushort_t* krow = kb + ((size_t)bh * S_ + j0 + lm) * DH_;
        b0.u[0] = *(const uint4*)(krow + kg * 16);
        b0.u[1] = *(const uint4*)(krow + kg * 16 + 8);
        b1.u[0] = *(const uint4*)(krow + 32 + kg * 16);
        b1.u[1] = *(const uint4*)(krow + 32 + kg * 16 + 8);
    };

    // ---- scores: S = q @ k^T, masked sliding window (pipelined) ----
    const int jt0 = (i0 >= WIN_) ? 0 : (WIN_ - i0) / 16;  // first in-seq tile
    Frag b0c, b1c, b0n, b1n;
    load_ktile(jbase + jt0 * 16, b0c, b1c);
    for (int jt = jt0; jt < NTILE_; ++jt) {
        const int j0 = jbase + jt * 16;
        if (jt + 1 < NTILE_) load_ktile(j0 + 16, b0n, b1n);  // prefetch
        v8f s8 = {};
        s8 = wmma_bf16(a0.v, b0c.v, s8);
        s8 = wmma_bf16(a1.v, b1c.v, s8);
        #pragma unroll
        for (int r = 0; r < 8; ++r) {
            const int m = r + kg * 8;
            const int i = i0 + m;
            const int j = j0 + lm;
            const bool valid = (j <= i) && (j >= i - (WIN_ - 1));
            sc[m * SCOLS_ + (j - jbase)] = valid ? s8[r] : -3.402823466e38f;
        }
        b0c = b0n;
        b1c = b1n;
    }
    __syncthreads();

    // ---- softmax over each row (2 lanes per row) ----
    {
        const int row = lane >> 1;
        const int half = lane & 1;
        float* prow = sc + row * SCOLS_ + half * (SVALID_ / 2);
        float mx = -3.402823466e38f;
        for (int c = 0; c < SVALID_ / 2; ++c) mx = fmaxf(mx, prow[c]);
        mx = fmaxf(mx, __shfl_xor(mx, 1));
        float sum = 0.0f;
        for (int c = 0; c < SVALID_ / 2; ++c) {
            const float p = __expf(prow[c] - mx);
            prow[c] = p;                       // unnormalized prob
            sum += p;
        }
        sum += __shfl_xor(sum, 1);
        if (!half) rowsum[row] = sum;
    }
    __syncthreads();

    // ---- O = P @ V  (17 K-chunks of 32 keys, 4 DH tiles) ----
    v8f o[4] = {};
    for (int kc = 0; kc < 17; ++kc) {
        const int cb = kc * 32;

        // V fragments first so the 8 loads clause together; the A-fragment
        // LDS reads + f2bf VALU below cover their latency.
        int ks = jbase + cb + kg * 16;         // 16-aligned key segment
        if (ks < 0 || ks + 16 > S_) ks = 0;    // fully-masked segment: P==0
        Frag fb[4];
        #pragma unroll
        for (int nt = 0; nt < 4; ++nt) {
            const ushort_t* vrow =
                vb + ((size_t)bh * DH_ + nt * 16 + lm) * S_ + ks;
            fb[nt].u[0] = *(const uint4*)(vrow);
            fb[nt].u[1] = *(const uint4*)(vrow + 8);
        }

        // A fragment from LDS probs (row = lm), fp32 -> bf16
        const float* pr = sc + lm * SCOLS_;
        const float4 f0 = *(const float4*)(pr + cb + kg * 16);
        const float4 f1 = *(const float4*)(pr + cb + kg * 16 + 4);
        const float4 f2 = *(const float4*)(pr + cb + kg * 16 + 8);
        const float4 f3 = *(const float4*)(pr + cb + kg * 16 + 12);
        Frag fa;
        fa.s[0]  = f2bf(f0.x); fa.s[1]  = f2bf(f0.y);
        fa.s[2]  = f2bf(f0.z); fa.s[3]  = f2bf(f0.w);
        fa.s[4]  = f2bf(f1.x); fa.s[5]  = f2bf(f1.y);
        fa.s[6]  = f2bf(f1.z); fa.s[7]  = f2bf(f1.w);
        fa.s[8]  = f2bf(f2.x); fa.s[9]  = f2bf(f2.y);
        fa.s[10] = f2bf(f2.z); fa.s[11] = f2bf(f2.w);
        fa.s[12] = f2bf(f3.x); fa.s[13] = f2bf(f3.y);
        fa.s[14] = f2bf(f3.z); fa.s[15] = f2bf(f3.w);

        #pragma unroll
        for (int nt = 0; nt < 4; ++nt)
            o[nt] = wmma_bf16(fa.v, fb[nt].v, o[nt]);
    }

    // ---- normalize and store attention output (B,S,H*DH) bf16 ----
    #pragma unroll
    for (int nt = 0; nt < 4; ++nt) {
        #pragma unroll
        for (int r = 0; r < 8; ++r) {
            const int m = r + kg * 8;
            const float res = o[nt][r] / rowsum[m];
            ab[((size_t)b * S_ + i0 + m) * (H_ * DH_) + h * DH_ + nt * 16 + lm] =
                f2bf(res);
        }
    }
}

// --------------------------- output projection -----------------------------
// d_out(M_,DIM_) fp32 = attn_bf(M_,1024) @ WoutT(1024,1024)^T

__global__ void __launch_bounds__(256)
gemm_out_kernel(const ushort_t* __restrict__ A,
                const ushort_t* __restrict__ BT,
                float* __restrict__ out) {
    __shared__ __align__(16) ushort_t sb[2 * BBUF_];
    const int bm   = blockIdx.x & 31;          // M/128 = 32
    const int nt4  = blockIdx.x >> 5;          // DIM/64 = 16
    const int lane = threadIdx.x & 31;
    const int lm   = lane & 15;
    const int kg   = lane >> 4;
    const int mrow = bm * 128 + (threadIdx.x >> 5) * 16;
    const int n0   = nt4 * 64;

    v8f acc[4] = {};
    gemm_core_128x64(A, BT, mrow, n0, sb, acc);

    #pragma unroll
    for (int ns = 0; ns < 4; ++ns) {
        const int n = n0 + ns * 16 + lm;
        #pragma unroll
        for (int r = 0; r < 8; ++r) {
            const int gm = mrow + r + kg * 8;
            out[(size_t)gm * DIM_ + n] = acc[ns][r];
        }
    }
}

// ------------------------------- launcher ----------------------------------

extern "C" void kernel_launch(void* const* d_in, const int* in_sizes, int n_in,
                              void* d_out, int out_size, void* d_ws, size_t ws_size,
                              hipStream_t stream) {
    (void)in_sizes; (void)n_in; (void)out_size; (void)ws_size;
    const float* x    = (const float*)d_in[0];   // (B,S,DIM)
    const float* Wqkv = (const float*)d_in[1];   // (DIM, 3*H*DH)
    const float* Wout = (const float*)d_in[2];   // (H*DH, DIM)
    float* out = (float*)d_out;

    // workspace layout (bytes)
    char* ws = (char*)d_ws;
    ushort_t* x_bf   = (ushort_t*)(ws);                          //  8 MiB
    ushort_t* WqkvT  = (ushort_t*)(ws + (8u << 20));             //  6 MiB
    ushort_t* WoutT  = (ushort_t*)(ws + (14u << 20));            //  2 MiB
    ushort_t* qbuf   = (ushort_t*)(ws + (16u << 20));            //  8 MiB
    ushort_t* kbuf   = (ushort_t*)(ws + (24u << 20));            //  8 MiB
    ushort_t* vbuf   = (ushort_t*)(ws + (32u << 20));            //  8 MiB
    ushort_t* abuf   = (ushort_t*)(ws + (40u << 20));            //  8 MiB

    // 1) fp32 -> bf16 conversions (weights transposed to (N,K))
    cvt_bf16_kernel<<<4096, 256, 0, stream>>>(x, x_bf, M_ * DIM_);
    cvt_bf16_t_kernel<<<3072, 256, 0, stream>>>(Wqkv, WqkvT, DIM_, N3_);
    cvt_bf16_t_kernel<<<1024, 256, 0, stream>>>(Wout, WoutT, H_ * DH_, DIM_);

    // 2) QKV projection GEMM (block = 128x64 tile, LDS-staged weights)
    gemm_qkv_kernel<<<32 * (N3_ / 64), 256, 0, stream>>>(x_bf, WqkvT,
                                                         qbuf, kbuf, vbuf);

    // 3) sliding-window attention (one wave per (b,h,16-query tile))
    attn_kernel<<<B_ * H_ * (S_ / 16), 32, 0, stream>>>(qbuf, kbuf, vbuf, abuf);

    // 4) output projection GEMM -> fp32 d_out
    gemm_out_kernel<<<32 * (DIM_ / 64), 256, 0, stream>>>(abuf, WoutT, out);
}
